// GCNGeneratorEncoder_7533372637745
// MI455X (gfx1250) — matrix operand order repaired and checked by reading, not compiled, not run on hardware
//
#include <hip/hip_runtime.h>
#include <hip/hip_bf16.h>

typedef __attribute__((ext_vector_type(16))) _Float16 v16h;
typedef __attribute__((ext_vector_type(8)))  float    v8f;

// ---------------------------------------------------------------------------
// Weight pre-pack: W[K x COLS] f32 -> Wp in per-lane WMMA B layout, f16,
// K zero-padded to KT*32. Element (kc, t, lane, e) holds
//   B[k][n], k = kc*32 + 16*(e>=8) + 8*(lane>=16) + (e&7), n = t*16 + (lane&15)
// so each lane's 16 halves for a (kc,t) tile are 32B contiguous.
// ---------------------------------------------------------------------------
template <int K, int COLS>
__global__ void pack_w_kernel(const float* __restrict__ W,
                              _Float16* __restrict__ Wp)
{
    constexpr int KT = (K + 31) / 32;
    constexpr int CT = COLS / 16;
    constexpr int total = KT * CT * 32 * 16;
    int idx = blockIdx.x * blockDim.x + threadIdx.x;
    if (idx >= total) return;
    const int e    = idx & 15;
    const int lane = (idx >> 4) & 31;
    const int rest = idx >> 9;
    const int t    = rest % CT;
    const int kc   = rest / CT;
    const int k    = kc * 32 + ((e >> 3) ? 16 : 0) + 8 * (lane >> 4) + (e & 7);
    const int n    = t * 16 + (lane & 15);
    Wp[idx] = (k < K) ? (_Float16)W[k * COLS + n] : (_Float16)0.f;
}

// ---------------------------------------------------------------------------
// GEMM: H[nrows x COLS] = X[nrows x K] @ W[K x COLS], f16 WMMA, f32 accum.
// One wave computes a 16-row strip across all COLS (CT tiles of 16 cols),
// KT = ceil(K/32) k-chunks of V_WMMA_F32_16X16X32_F16.
// A: unconditional float4 loads + cvt_pk (range predicates compile-time).
// B: pre-packed, one contiguous 32B v16h per lane per tile.
// C: wave-uniform full-tile test -> unguarded stores w/ immediate offsets.
// ---------------------------------------------------------------------------
template <int K, int COLS>
__global__ void gemm_wmma_kernel(const float* __restrict__ X,
                                 const _Float16* __restrict__ Wp,
                                 float* __restrict__ H, int nrows)
{
    static_assert(K % 16 == 0, "K multiple of 16");
    constexpr int KT = (K + 31) / 32;
    constexpr int CT = COLS / 16;

    const int wave = blockIdx.x * (blockDim.x >> 5) + (threadIdx.x >> 5);
    const int lane = threadIdx.x & 31;
    const int row0 = wave << 4;
    if (row0 >= nrows) return;            // wave-uniform: EXEC stays all-1s

    const int hs = lane >> 4;             // half-wave select
    const int lm = lane & 15;             // M (for A) / N (for B/C/D)

    v8f acc[CT] = {};

    const int arow = (row0 + lm < nrows) ? (row0 + lm) : (nrows - 1);
    const float* xrow = X + (size_t)arow * K;

#pragma unroll
    for (int kc = 0; kc < KT; ++kc) {
        const int kbase = kc * 32;
        // A tile 16x32 f16 (ISA 7.12.2): lane half 'hs' holds
        //   K = kbase + 8*hs + [0..7]      (elements 0..7)
        //   K = kbase + 16 + 8*hs + [0..7] (elements 8..15)
        v16h a;
        {
            const float* p = xrow + kbase + hs * 8;       // 32B aligned
            const float4 q0 = *(const float4*)(p);
            const float4 q1 = *(const float4*)(p + 4);
            a[0] = (_Float16)q0.x; a[1] = (_Float16)q0.y;
            a[2] = (_Float16)q0.z; a[3] = (_Float16)q0.w;
            a[4] = (_Float16)q1.x; a[5] = (_Float16)q1.y;
            a[6] = (_Float16)q1.z; a[7] = (_Float16)q1.w;
        }
        if (kbase + 32 <= K) {            // compile-time after unroll
            const float* p = xrow + kbase + 16 + hs * 8;
            const float4 q0 = *(const float4*)(p);
            const float4 q1 = *(const float4*)(p + 4);
            a[8]  = (_Float16)q0.x; a[9]  = (_Float16)q0.y;
            a[10] = (_Float16)q0.z; a[11] = (_Float16)q0.w;
            a[12] = (_Float16)q1.x; a[13] = (_Float16)q1.y;
            a[14] = (_Float16)q1.z; a[15] = (_Float16)q1.w;
        } else {
            a[8]  = (_Float16)0.f; a[9]  = (_Float16)0.f;
            a[10] = (_Float16)0.f; a[11] = (_Float16)0.f;
            a[12] = (_Float16)0.f; a[13] = (_Float16)0.f;
            a[14] = (_Float16)0.f; a[15] = (_Float16)0.f;
        }
#pragma unroll
        for (int t = 0; t < CT; ++t) {
            const v16h b = *(const v16h*)(
                Wp + ((size_t)(kc * CT + t) * 32 + lane) * 16);
            acc[t] = __builtin_amdgcn_wmma_f32_16x16x32_f16(
                false, a, false, b, (short)0, acc[t], false, false);
        }
    }

    // C/D layout: VGPR r holds row M = r + 8*hs, column N = lm.
    // Wave-uniform full-tile test keeps the hot path free of EXEC juggling;
    // offsets r*COLS + t*16 become instruction-immediate offsets.
    float* hb = H + (size_t)(row0 + 8 * hs) * COLS + lm;
    if (row0 + 16 <= nrows) {
#pragma unroll
        for (int t = 0; t < CT; ++t)
#pragma unroll
            for (int r = 0; r < 8; ++r)
                hb[r * COLS + t * 16] = acc[t][r];
    } else {
#pragma unroll
        for (int t = 0; t < CT; ++t)
#pragma unroll
            for (int r = 0; r < 8; ++r)
                if (row0 + 8 * hs + r < nrows)
                    hb[r * COLS + t * 16] = acc[t][r];
    }
}

// ---------------------------------------------------------------------------
// Helper kernels
// ---------------------------------------------------------------------------
__global__ void fill_kernel(float* __restrict__ p, float v, long long n)
{
    long long i = (long long)blockIdx.x * blockDim.x + threadIdx.x;
    if (i < n) p[i] = v;
}

__global__ void deg_edges_kernel(const int* __restrict__ dst,
                                 float* __restrict__ deg, int nE)
{
    int e = blockIdx.x * blockDim.x + threadIdx.x;
    if (e < nE) atomicAdd(&deg[dst[e]], 1.0f);
}

__global__ void dinv_kernel(const float* __restrict__ deg,
                            float* __restrict__ dinv, int n)
{
    int i = blockIdx.x * blockDim.x + threadIdx.x;
    if (i < n) {
        float d = deg[i];
        dinv[i] = (d > 0.f) ? rsqrtf(d) : 0.f;
    }
}

// Edge scatter-add: out[dst] += H[src] * dinv[src]*dinv[dst].
// C/4 threads per edge, float4 payload, f32 atomics resolving in L2
// (N*64*4B = 12.8MB output lives in the 192MB L2).
template <int C>
__global__ void scatter_kernel(const int* __restrict__ src,
                               const int* __restrict__ dst,
                               const float* __restrict__ dinv,
                               const float* __restrict__ H,
                               float* __restrict__ agg, int nE)
{
    constexpr int TPE = C / 4;
    long long tid = (long long)blockIdx.x * blockDim.x + threadIdx.x;
    int e = (int)(tid / TPE);
    int q = (int)(tid % TPE);
    if (e >= nE) return;
    const int s = src[e];
    const int d = dst[e];
    const float w = dinv[s] * dinv[d];
    const float4 h = *(const float4*)(H + (size_t)s * C + q * 4);
    float* o = agg + (size_t)d * C + q * 4;
    atomicAdd(o + 0, h.x * w);
    atomicAdd(o + 1, h.y * w);
    atomicAdd(o + 2, h.z * w);
    atomicAdd(o + 3, h.w * w);
}

// Self-loop (norm = dinv[i]^2) + bias + activation. ACT: 0=leaky_relu(0.2), 1=tanh
template <int C, int ACT>
__global__ void post_kernel(const float* __restrict__ agg,
                            const float* __restrict__ H,
                            const float* __restrict__ dinv,
                            const float* __restrict__ bias,
                            float* __restrict__ out, int n)
{
    constexpr int QP = C / 4;
    long long tid = (long long)blockIdx.x * blockDim.x + threadIdx.x;
    if (tid >= (long long)n * QP) return;
    const int node = (int)(tid / QP);
    const int q    = (int)(tid % QP);
    float w = dinv[node]; w *= w;
    const float4 a = ((const float4*)agg)[tid];
    const float4 h = ((const float4*)H)[tid];
    const float4 b = *(const float4*)(bias + q * 4);
    float4 v;
    v.x = a.x + h.x * w + b.x;
    v.y = a.y + h.y * w + b.y;
    v.z = a.z + h.z * w + b.z;
    v.w = a.w + h.w * w + b.w;
    if (ACT == 0) {
        v.x = v.x > 0.f ? v.x : 0.2f * v.x;
        v.y = v.y > 0.f ? v.y : 0.2f * v.y;
        v.z = v.z > 0.f ? v.z : 0.2f * v.z;
        v.w = v.w > 0.f ? v.w : 0.2f * v.w;
    } else {
        v.x = tanhf(v.x); v.y = tanhf(v.y);
        v.z = tanhf(v.z); v.w = tanhf(v.w);
    }
    ((float4*)out)[tid] = v;
}

// ---------------------------------------------------------------------------
// Launch
// ---------------------------------------------------------------------------
extern "C" void kernel_launch(void* const* d_in, const int* in_sizes, int n_in,
                              void* d_out, int out_size, void* d_ws, size_t ws_size,
                              hipStream_t stream)
{
    const float* x  = (const float*)d_in[0];
    const int*   ei = (const int*)d_in[1];
    const float* W1 = (const float*)d_in[2];
    const float* b1 = (const float*)d_in[3];
    const float* W2 = (const float*)d_in[4];
    const float* b2 = (const float*)d_in[5];
    const float* W3 = (const float*)d_in[6];
    const float* b3 = (const float*)d_in[7];
    float* out = (float*)d_out;

    const int N = in_sizes[0] / 16;
    const int E = in_sizes[1] / 2;
    const int* src = ei;
    const int* dst = ei + E;

    // workspace layout (floats; every region a multiple of 32B)
    float* ws   = (float*)d_ws;
    float* deg  = ws;                        // N
    float* dinv = deg + N;                   // N
    float* buf0 = dinv + N;                  // N*64  (gemm output)
    float* buf1 = buf0 + (size_t)N * 64;     // N*64  (edge aggregation)
    float* buf2 = buf1 + (size_t)N * 64;     // N*64  (activated features)
    _Float16* w1p = (_Float16*)(buf2 + (size_t)N * 64);  // 1*4*512 = 2048 halves
    _Float16* w2p = w1p + 2048;                          // 2*4*512 = 4096 halves
    _Float16* w3p = w2p + 4096;                          // 2*1*512 = 1024 halves

    const int T = 256;
    const int gN    = (N + T - 1) / T;
    const int gE    = (E + T - 1) / T;
    const int gemmB = ((N + 15) / 16 + 7) / 8;     // 8 waves/block, 16 rows/wave
    const long long n64  = (long long)N * 64;
    const long long n16  = (long long)N * 16;
    const int g64   = (int)((n64 + T - 1) / T);
    const int g16   = (int)((n16 + T - 1) / T);
    const int gS64  = (int)(((long long)E * 16 + T - 1) / T);  // 16 thr/edge
    const int gS16  = (int)(((long long)E * 4  + T - 1) / T);  // 4  thr/edge
    const int gP64  = (int)(((long long)N * 16 + T - 1) / T);  // N*C/4 threads
    const int gP16  = (int)(((long long)N * 4  + T - 1) / T);

    // --- pre-pack weights into WMMA B layout (tiny, one-time per launch)
    pack_w_kernel<16, 64><<<(2048 + T - 1) / T, T, 0, stream>>>(W1, w1p);
    pack_w_kernel<64, 64><<<(4096 + T - 1) / T, T, 0, stream>>>(W2, w2p);
    pack_w_kernel<64, 16><<<(1024 + T - 1) / T, T, 0, stream>>>(W3, w3p);

    // --- gcn_norm: deg (with self-loop) -> dinv = rsqrt(deg)
    fill_kernel<<<gN, T, 0, stream>>>(deg, 1.0f, N);
    deg_edges_kernel<<<gE, T, 0, stream>>>(dst, deg, E);
    dinv_kernel<<<gN, T, 0, stream>>>(deg, dinv, N);

    // --- layer 1: H1 = x @ W1 ; agg ; leaky_relu
    gemm_wmma_kernel<16, 64><<<gemmB, T, 0, stream>>>(x, w1p, buf0, N);
    fill_kernel<<<g64, T, 0, stream>>>(buf1, 0.0f, n64);
    scatter_kernel<64><<<gS64, T, 0, stream>>>(src, dst, dinv, buf0, buf1, E);
    post_kernel<64, 0><<<gP64, T, 0, stream>>>(buf1, buf0, dinv, b1, buf2, N);

    // --- layer 2: H2 = h @ W2 ; agg ; leaky_relu
    gemm_wmma_kernel<64, 64><<<gemmB, T, 0, stream>>>(buf2, w2p, buf0, N);
    fill_kernel<<<g64, T, 0, stream>>>(buf1, 0.0f, n64);
    scatter_kernel<64><<<gS64, T, 0, stream>>>(src, dst, dinv, buf0, buf1, E);
    post_kernel<64, 0><<<gP64, T, 0, stream>>>(buf1, buf0, dinv, b2, buf2, N);

    // --- layer 3: H3 = h @ W3 ; agg ; tanh -> out
    gemm_wmma_kernel<64, 16><<<gemmB, T, 0, stream>>>(buf2, w3p, buf0, N);
    fill_kernel<<<g16, T, 0, stream>>>(buf1, 0.0f, n16);
    scatter_kernel<16><<<gS16, T, 0, stream>>>(src, dst, dinv, buf0, buf1, E);
    post_kernel<16, 1><<<gP16, T, 0, stream>>>(buf1, buf0, dinv, b3, out, N);
}
